// MultiHeadAttentionBlock_7009386627262
// MI455X (gfx1250) — matrix operand order repaired and checked by reading, not compiled, not run on hardware
//
#include <hip/hip_runtime.h>

// ---------------------------------------------------------------------------
// MHA block for MI455X (gfx1250, wave32): bf16 WMMA + async LDS staging.
// ---------------------------------------------------------------------------

typedef __attribute__((ext_vector_type(16))) __bf16        v16bf;
typedef __attribute__((ext_vector_type(8)))  float         v8f;
typedef __attribute__((ext_vector_type(4)))  unsigned int  u32x4;
typedef __attribute__((ext_vector_type(4)))  int           i32x4;

// exact parameter types of __builtin_amdgcn_global_load_async_to_lds_b128:
// (int4 __device__ *, int4 __shared__ *, imm offset, imm cpol)
typedef __attribute__((address_space(1))) i32x4 gi32x4;
typedef __attribute__((address_space(3))) i32x4 li32x4;

union FragBF16 { u32x4 q[2]; v16bf v; };

constexpr int   Bb   = 2;
constexpr int   Ss   = 2048;
constexpr int   Ee   = 2048;
constexpr int   Hh   = 16;
constexpr int   DK   = 128;
constexpr int   Mm   = Bb * Ss;                 // 4096 flattened rows
constexpr float SCALE  = 0.08838834764831845f;  // 1/sqrt(128)
constexpr float NEGINF = -1.0e9f;

// ---- gfx1250 async global->LDS path (guarded; falls back to sync copies) ---
#if defined(__has_builtin)
#if __has_builtin(__builtin_amdgcn_global_load_async_to_lds_b128)
#define HAVE_ASYNC_LDS 1
#endif
#if __has_builtin(__builtin_amdgcn_s_wait_asynccnt)
#define HAVE_WAIT_ASYNC 1
#endif
#endif
#ifndef HAVE_ASYNC_LDS
#define HAVE_ASYNC_LDS 0
#endif
#ifndef HAVE_WAIT_ASYNC
#define HAVE_WAIT_ASYNC 0
#endif

template <int N>
__device__ __forceinline__ void wait_async() {
#if HAVE_WAIT_ASYNC
  __builtin_amdgcn_s_wait_asynccnt(N);
#else
  asm volatile("s_wait_asynccnt %0" ::"n"(N) : "memory");
#endif
}

// Copy one 32x128 bf16 K-tile (8 KB) into a per-wave LDS buffer: 16 async
// b128 transfers (each lane moves 16 B per instruction, wave moves 512 B).
__device__ __forceinline__ void stage_k_tile(const unsigned short* __restrict__ gsrc,
                                             unsigned short* __restrict__ ldst,
                                             int lane) {
#pragma unroll
  for (int i = 0; i < 16; ++i) {
    const unsigned short* g = gsrc + i * 256 + lane * 8;
    unsigned short*       l = ldst + i * 256 + lane * 8;
#if HAVE_ASYNC_LDS
    __builtin_amdgcn_global_load_async_to_lds_b128(
        (gi32x4*)g, (li32x4*)l, 0, 0);
#else
    *(u32x4*)l = *(const u32x4*)g;
#endif
  }
}

__device__ __forceinline__ unsigned short f2bf(float f) {
  unsigned int u = __float_as_uint(f);
  u += 0x7FFFu + ((u >> 16) & 1u);              // round-to-nearest-even
  return (unsigned short)(u >> 16);
}

// ---------------------------------------------------------------------------
// fp32 -> bf16 bulk conversion (float4 in, uint2 out)
// ---------------------------------------------------------------------------
__global__ void __launch_bounds__(256)
f32_to_bf16_kernel(const float* __restrict__ in, unsigned short* __restrict__ out, int n4) {
  int i = blockIdx.x * blockDim.x + threadIdx.x;
  if (i >= n4) return;
  float4 f = ((const float4*)in)[i];
  uint2 p;
  p.x = (unsigned)f2bf(f.x) | ((unsigned)f2bf(f.y) << 16);
  p.y = (unsigned)f2bf(f.z) | ((unsigned)f2bf(f.w) << 16);
  ((uint2*)out)[i] = p;
}

// ---------------------------------------------------------------------------
// Y = X (M x K, bf16 row-major) @ W^T (W is N x K, bf16 row-major)
// One wave computes a 32(M) x 64(N) tile: 2 A-frags, 4 B-frags, 8 f32 accs
// -> 8 WMMA per 12 b128 loads per K-chunk.
// Fragment addressing per CDNA5 ISA 16-bit layouts:
//   A 16x32: lane l holds row (l&15); K = (j>>2)*16 + (l>>4)*8 + (j&3)*2
//            -> two b128 loads at +khalf*8 and +16+khalf*8
//   B 32x16: lane l holds col (l&15); K = (j>>2)*16 + (l>>4)*8 + (j&3)*2 (same as A)
//            -> two b128 loads at +khalf*8 and +16+khalf*8
// MODE 0: bf16 out in (B,H,S,DK)   (Q / K projections)
// MODE 1: bf16 out in (B,H,DK,S)   (V projection, pre-transposed for P*V)
// MODE 2: fp32 out, plain (M,N)    (output projection -> d_out)
// ---------------------------------------------------------------------------
template <int MODE>
__global__ void __launch_bounds__(256)
gemm_xwt_wmma(const unsigned short* __restrict__ X,
              const unsigned short* __restrict__ W,
              void* __restrict__ Y) {
  const int lane  = threadIdx.x & 31;
  const int wave  = (blockIdx.x * 256 + threadIdx.x) >> 5;
  const int nstr  = Ee / 64;                    // 32 N-strips
  const int mt    = wave / nstr;                // 0..Mm/32-1
  const int n0    = (wave % nstr) * 64;
  const int khalf = lane >> 4;
  const int l15   = lane & 15;

  const unsigned short* pA0 = X + (size_t)(mt * 32 + l15) * Ee + khalf * 8;
  const unsigned short* pA1 = pA0 + (size_t)16 * Ee;

  v8f acc[2][4] = {};
  for (int k0 = 0; k0 < Ee; k0 += 32) {
    __builtin_prefetch(pA0 + k0 + 128, 0, 3);   // global_prefetch_b8
    __builtin_prefetch(pA1 + k0 + 128, 0, 3);
    FragBF16 a0, a1;
    a0.q[0] = *(const u32x4*)(pA0 + k0);
    a0.q[1] = *(const u32x4*)(pA0 + k0 + 16);
    a1.q[0] = *(const u32x4*)(pA1 + k0);
    a1.q[1] = *(const u32x4*)(pA1 + k0 + 16);
#pragma unroll
    for (int t = 0; t < 4; ++t) {
      const unsigned short* pB = W + (size_t)(n0 + t * 16 + l15) * Ee + k0 + khalf * 8;
      FragBF16 b;
      b.q[0] = *(const u32x4*)(pB);
      b.q[1] = *(const u32x4*)(pB + 16);
      acc[0][t] = __builtin_amdgcn_wmma_f32_16x16x32_bf16(
          false, a0.v, false, b.v, (short)0, acc[0][t], false, false);
      acc[1][t] = __builtin_amdgcn_wmma_f32_16x16x32_bf16(
          false, a1.v, false, b.v, (short)0, acc[1][t], false, false);
    }
  }

#pragma unroll
  for (int u = 0; u < 2; ++u) {
#pragma unroll
    for (int t = 0; t < 4; ++t) {
      const int n = n0 + t * 16 + l15;
#pragma unroll
      for (int i = 0; i < 8; ++i) {
        const int   m   = mt * 32 + u * 16 + i + 8 * khalf;
        const float val = acc[u][t][i];
        if (MODE == 2) {
          ((float*)Y)[(size_t)m * Ee + n] = val;
        } else {
          const int bb = m >> 11, s = m & (Ss - 1);
          const int h  = n >> 7,  d = n & (DK - 1);
          const size_t idx = (MODE == 0)
              ? ((size_t)((bb * Hh + h) * Ss + s)) * DK + d
              : ((size_t)((bb * Hh + h) * DK + d)) * Ss + s;
          ((unsigned short*)Y)[idx] = f2bf(val);
        }
      }
    }
  }
}

// ---------------------------------------------------------------------------
// Flash attention: one wave per (b, h, 16-query tile), online softmax.
// K-tiles are double-buffered through per-wave LDS via async global->LDS
// copies (ASYNCcnt pipelining); B-fragments then come from ds_load_b128.
// V is pre-transposed so its fragments are direct contiguous b128 loads.
// 2 waves per block -> 34 KB static LDS per block.
// ---------------------------------------------------------------------------
constexpr int AW = 2;  // waves per attention block

__global__ void __launch_bounds__(AW * 32)
flash_attn_wmma(const unsigned short* __restrict__ Qh,
                const unsigned short* __restrict__ Kh,
                const unsigned short* __restrict__ Vt,
                const unsigned char*  __restrict__ causal_mask,
                const unsigned char*  __restrict__ key_mask,
                unsigned short* __restrict__ AO) {
  __shared__ __align__(16) unsigned short ldsK[AW][2][32 * DK];  // 16 KB/wave
  __shared__ __align__(16) unsigned short ldsP[AW][16 * 32];     //  1 KB/wave

  const int lane  = threadIdx.x & 31;
  const int wib   = threadIdx.x >> 5;
  const int wave  = blockIdx.x * AW + wib;
  const int qt    = wave & (Ss / 16 - 1);       // 0..127
  const int bh    = wave >> 7;                  // 0..31
  const int b     = bh >> 4, h = bh & 15;
  const int khalf = lane >> 4;
  const int l15   = lane & 15;
  const int q0    = qt * 16;

  const unsigned short* Qp = Qh + (size_t)bh * Ss * DK;
  const unsigned short* Kp = Kh + (size_t)bh * Ss * DK;
  const unsigned short* Vp = Vt + (size_t)bh * DK * Ss;

  // Q tile: 16 x 128 = 4 A-fragments of 16x32
  FragBF16 qf[4];
  {
    const unsigned short* p = Qp + (size_t)(q0 + l15) * DK + khalf * 16;
#pragma unroll
    for (int c = 0; c < 4; ++c) {
      qf[c].q[0] = *(const u32x4*)(p + c * 32);
      qf[c].q[1] = *(const u32x4*)(p + c * 32 + 8);
    }
  }

  float mrow[8], lrow[8];
  v8f   o[8] = {};
#pragma unroll
  for (int i = 0; i < 8; ++i) { mrow[i] = -1e30f; lrow[i] = 0.0f; }

  unsigned short* pm = &ldsP[wib][0];

  // preload K tile 0
  stage_k_tile(Kp, &ldsK[wib][0][0], lane);

  for (int kk0 = 0; kk0 < Ss; kk0 += 32) {
    const int cur = (kk0 >> 5) & 1;
    // issue async copy of next tile, then wait for the current one
    if (kk0 + 32 < Ss) {
      stage_k_tile(Kp + (size_t)(kk0 + 32) * DK, &ldsK[wib][cur ^ 1][0], lane);
      wait_async<16>();   // async loads complete in order -> tile `cur` ready
    } else {
      wait_async<0>();
    }
    const unsigned short* kt = &ldsK[wib][cur][0];  // [kk_rel][d] row-major

    // ---- scores: S = Q K^T (16x32 tile, two 16x16 halves) ----
    v8f slo = {}, shi = {};
    {
      const unsigned short* pk = kt + l15 * DK + khalf * 16;         // kk = l15
#pragma unroll
      for (int c = 0; c < 4; ++c) {
        FragBF16 kb;
        kb.q[0] = *(const u32x4*)(pk + c * 32);                      // ds_load_b128
        kb.q[1] = *(const u32x4*)(pk + c * 32 + 8);
        slo = __builtin_amdgcn_wmma_f32_16x16x32_bf16(
            false, qf[c].v, false, kb.v, (short)0, slo, false, false);
      }
      const unsigned short* pk2 = kt + (16 + l15) * DK + khalf * 16; // kk = 16+l15
#pragma unroll
      for (int c = 0; c < 4; ++c) {
        FragBF16 kb;
        kb.q[0] = *(const u32x4*)(pk2 + c * 32);
        kb.q[1] = *(const u32x4*)(pk2 + c * 32 + 8);
        shi = __builtin_amdgcn_wmma_f32_16x16x32_bf16(
            false, qf[c].v, false, kb.v, (short)0, shi, false, false);
      }
    }

    // ---- scale + pre-softmax key (padding) mask ----
    const bool kmlo = key_mask[b * Ss + kk0 + l15] != 0;
    const bool kmhi = key_mask[b * Ss + kk0 + 16 + l15] != 0;
#pragma unroll
    for (int i = 0; i < 8; ++i) {
      slo[i] = kmlo ? NEGINF : slo[i] * SCALE;
      shi[i] = kmhi ? NEGINF : shi[i] * SCALE;
    }

    // ---- online softmax, per row (half-wave shfl_xor reductions;
    //      rows 0-7 live in lanes 0-15, rows 8-15 in lanes 16-31) ----
#pragma unroll
    for (int i = 0; i < 8; ++i) {
      float t = fmaxf(slo[i], shi[i]);
#pragma unroll
      for (int msk = 1; msk < 16; msk <<= 1) t = fmaxf(t, __shfl_xor(t, msk, 32));
      const float mnew  = fmaxf(mrow[i], t);
      const float alpha = __expf(mrow[i] - mnew);
      const float plo   = __expf(slo[i] - mnew);
      const float phi   = __expf(shi[i] - mnew);
      float rs = plo + phi;
#pragma unroll
      for (int msk = 1; msk < 16; msk <<= 1) rs += __shfl_xor(rs, msk, 32);
      lrow[i] = lrow[i] * alpha + rs;
      mrow[i] = mnew;
#pragma unroll
      for (int c = 0; c < 8; ++c) o[c][i] *= alpha;
      // stash P (bf16) into per-wave LDS tile (C-layout -> row-major)
      const int row = i + 8 * khalf;
      pm[row * 32 + l15]      = f2bf(plo);
      pm[row * 32 + 16 + l15] = f2bf(phi);
    }

    // ---- reload P as a 16x32 A-fragment (layout change via LDS) ----
    FragBF16 pf;
    pf.q[0] = *(const u32x4*)(pm + l15 * 32 + khalf * 16);
    pf.q[1] = *(const u32x4*)(pm + l15 * 32 + khalf * 16 + 8);

    // ---- O += P * V  (V pre-transposed: contiguous b128 B-frag loads) ----
#pragma unroll
    for (int c = 0; c < 8; ++c) {
      const unsigned short* pv = Vp + (size_t)(c * 16 + l15) * Ss + kk0 + khalf * 16;
      FragBF16 vb;
      vb.q[0] = *(const u32x4*)(pv);
      vb.q[1] = *(const u32x4*)(pv + 8);
      o[c] = __builtin_amdgcn_wmma_f32_16x16x32_bf16(
          false, pf.v, false, vb.v, (short)0, o[c], false, false);
    }
  }

  // ---- epilogue: 1/l, post-softmax causal row zeroing, bf16 store ----
#pragma unroll
  for (int i = 0; i < 8; ++i) {
    const int qrow = q0 + i + 8 * khalf;
    float inv = 1.0f / lrow[i];
    if (causal_mask[b * Ss + qrow]) inv = 0.0f;
    unsigned short* po = AO + ((size_t)(b * Ss + qrow)) * Ee + h * DK + l15;
#pragma unroll
    for (int c = 0; c < 8; ++c) po[c * 16] = f2bf(o[c][i] * inv);
  }
}

// ---------------------------------------------------------------------------
// Host-side orchestration
// ---------------------------------------------------------------------------
extern "C" void kernel_launch(void* const* d_in, const int* in_sizes, int n_in,
                              void* d_out, int out_size, void* d_ws, size_t ws_size,
                              hipStream_t stream) {
  (void)in_sizes; (void)n_in; (void)out_size; (void)ws_size;

  const float* q  = (const float*)d_in[0];
  const float* k  = (const float*)d_in[1];
  const float* v  = (const float*)d_in[2];
  const unsigned char* causal = (const unsigned char*)d_in[3];
  const unsigned char* keym   = (const unsigned char*)d_in[4];
  const float* wq = (const float*)d_in[5];
  const float* wk = (const float*)d_in[6];
  const float* wv = (const float*)d_in[7];
  const float* wo = (const float*)d_in[8];
  float* out = (float*)d_out;

  unsigned short* ws = (unsigned short*)d_ws;
  const size_t SZ_ACT = (size_t)Mm * Ee;   // 8,388,608 elems
  const size_t SZ_W   = (size_t)Ee * Ee;   // 4,194,304 elems

  unsigned short* qb  = ws;                  // bf16 activations
  unsigned short* kb  = qb  + SZ_ACT;
  unsigned short* vb  = kb  + SZ_ACT;
  unsigned short* wqb = vb  + SZ_ACT;        // bf16 weights
  unsigned short* wkb = wqb + SZ_W;
  unsigned short* wvb = wkb + SZ_W;
  unsigned short* wob = wvb + SZ_W;
  unsigned short* Qh  = wob + SZ_W;          // (B,H,S,DK)
  unsigned short* Kh  = Qh  + SZ_ACT;
  unsigned short* Vt  = Kh  + SZ_ACT;        // (B,H,DK,S)
  unsigned short* ao  = Vt  + SZ_ACT;        // attention output (B,S,E)

  // 1) fp32 -> bf16 conversions
  {
    const int n4a = (int)(SZ_ACT / 4), n4w = (int)(SZ_W / 4);
    const dim3 blk(256);
    f32_to_bf16_kernel<<<dim3((n4a + 255) / 256), blk, 0, stream>>>(q,  qb,  n4a);
    f32_to_bf16_kernel<<<dim3((n4a + 255) / 256), blk, 0, stream>>>(k,  kb,  n4a);
    f32_to_bf16_kernel<<<dim3((n4a + 255) / 256), blk, 0, stream>>>(v,  vb,  n4a);
    f32_to_bf16_kernel<<<dim3((n4w + 255) / 256), blk, 0, stream>>>(wq, wqb, n4w);
    f32_to_bf16_kernel<<<dim3((n4w + 255) / 256), blk, 0, stream>>>(wk, wkb, n4w);
    f32_to_bf16_kernel<<<dim3((n4w + 255) / 256), blk, 0, stream>>>(wv, wvb, n4w);
    f32_to_bf16_kernel<<<dim3((n4w + 255) / 256), blk, 0, stream>>>(wo, wob, n4w);
  }

  // 2) projections: waves = (M/32)*(N/64) = 4096 -> 512 blocks x 256 thr
  const dim3 gemm_grid(512), gemm_blk(256);
  gemm_xwt_wmma<0><<<gemm_grid, gemm_blk, 0, stream>>>(qb, wqb, Qh);
  gemm_xwt_wmma<0><<<gemm_grid, gemm_blk, 0, stream>>>(kb, wkb, Kh);
  gemm_xwt_wmma<1><<<gemm_grid, gemm_blk, 0, stream>>>(vb, wvb, Vt);

  // 3) attention: waves = B*H*(S/16) = 4096 -> 2048 blocks x 2 waves
  flash_attn_wmma<<<dim3(4096 / AW), dim3(AW * 32), 0, stream>>>(
      Qh, Kh, Vt, causal, keym, ao);

  // 4) output projection -> fp32 d_out
  gemm_xwt_wmma<2><<<gemm_grid, gemm_blk, 0, stream>>>(ao, wob, out);
}
